// MCSModule_57526791962872
// MI455X (gfx1250) — matrix operand order repaired and checked by reading, not compiled, not run on hardware
//
#include <hip/hip_runtime.h>
#include <hip/hip_bf16.h>
#include <math.h>

#define NGRAPH 64      // G
#define NPG    128     // nodes per graph
#define DF     64      // feature dim
#define NNODE  (NGRAPH*NPG)
#define NEDGE  131072

typedef __attribute__((ext_vector_type(16))) _Float16 v16h;
typedef __attribute__((ext_vector_type(2)))  _Float16 v2h;
typedef __attribute__((ext_vector_type(8)))  float    v8f;

// order-preserving float <-> uint key so we can use integer atomicMax
__device__ __forceinline__ unsigned fkey(float x) {
    unsigned b = __float_as_uint(x);
    return (b & 0x80000000u) ? ~b : (b | 0x80000000u);
}
__device__ __forceinline__ float fdec(unsigned k) {
    return __uint_as_float((k & 0x80000000u) ? (k ^ 0x80000000u) : ~k);
}

__global__ void zero_deg_kernel(float* deg) {
    int i = blockIdx.x * blockDim.x + threadIdx.x;
    if (i < NNODE) deg[i] = 0.0f;
}

__global__ void accum_deg_kernel(const int* __restrict__ ei, float* __restrict__ deg) {
    int e = blockIdx.x * blockDim.x + threadIdx.x;
    if (e < NEDGE) {
        atomicAdd(&deg[ei[e]],         1.0f);   // edge_index[0, e]
        atomicAdd(&deg[ei[NEDGE + e]], 1.0f);   // edge_index[1, e]
    }
}

// One block per (g,h) graph pair. 256 threads = 8 waves.
// Wave w computes output rows [16w, 16w+16) of the 128x128 sim block.
// WMMA computes D = Xg_tile * Xh_tile^T + C, with
//   C[a,b] = -0.5*(|x_a|^2 + |x_b|^2 + (deg_a - deg_b)^2)
// so t = d2 + dd = -2*D and max_b sim = exp(2 * max_b D).
__global__ void __launch_bounds__(256) mcs_pair_kernel(const float* __restrict__ x,
                                                       const float* __restrict__ deg,
                                                       float* __restrict__ matchbuf) {
    __shared__ _Float16 Ag[NPG * DF] __attribute__((aligned(16)));
    __shared__ _Float16 Bh[NPG * DF] __attribute__((aligned(16)));
    __shared__ float sqA[NPG], sqB[NPG], dgA[NPG], dgB[NPG];
    __shared__ unsigned colKey[NPG];
    __shared__ float mAcc[2];   // [0] = m_row, [1] = m_col

    const int g = blockIdx.y, h = blockIdx.x;
    const int tid = threadIdx.x;
    const float* xg = x + (size_t)g * NPG * DF;
    const float* xh = x + (size_t)h * NPG * DF;

    // Stage both 128x64 tiles in LDS as f16 (coalesced dword loads).
    for (int i = tid; i < NPG * DF; i += 256) {
        Ag[i] = (_Float16)xg[i];
        Bh[i] = (_Float16)xh[i];
    }
    // Per-row squared norms + degrees (f32, matching the reference epilogue math).
    if (tid < NPG) {
        const float* p = xg + tid * DF;
        float s = 0.0f;
        #pragma unroll 8
        for (int k = 0; k < DF; ++k) s += p[k] * p[k];
        sqA[tid] = s;
        dgA[tid] = deg[g * NPG + tid];
        colKey[tid] = 0u;                 // smallest key == -inf sentinel
    } else {
        const int r = tid - NPG;
        const float* p = xh + r * DF;
        float s = 0.0f;
        #pragma unroll 8
        for (int k = 0; k < DF; ++k) s += p[k] * p[k];
        sqB[r] = s;
        dgB[r] = deg[h * NPG + r];
    }
    if (tid < 2) mAcc[tid] = 0.0f;
    __syncthreads();

    const int wave = tid >> 5;
    const int lane = tid & 31;
    const int hi   = lane >> 4;       // half-wave select
    const int ln   = lane & 15;
    const int rowBase = wave * 16;

    // A fragments (16x32 f16 per k-step): lanes 0-15 & 16-31 both carry M=ln,
    // K pairs per documented layout: j<4 -> K=2j(+8 for hi half), j>=4 -> K=2j+8(+8).
    v16h a0, a1;
    {
        const int arow = rowBase + ln;
        #pragma unroll
        for (int j = 0; j < 8; ++j) {
            const int k0 = ((j < 4) ? 2 * j : 2 * j + 8) + hi * 8;
            v2h p0 = *(const v2h*)&Ag[arow * DF + k0];
            v2h p1 = *(const v2h*)&Ag[arow * DF + 32 + k0];
            a0[2 * j] = p0.x; a0[2 * j + 1] = p0.y;
            a1[2 * j] = p1.x; a1[2 * j + 1] = p1.y;
        }
    }

    float rm[8];
    #pragma unroll
    for (int i = 0; i < 8; ++i) rm[i] = -3.4e38f;

    for (int tn = 0; tn < 8; ++tn) {
        const int b = tn * 16 + ln;   // column node in graph h
        // B fragments (32x16 f16 per k-step): same K layout as A: j<4 -> K=2j(+8 for hi half), j>=4 -> K=2j+8(+8).
        v16h b0, b1;
        #pragma unroll
        for (int j = 0; j < 8; ++j) {
            const int k0 = ((j < 4) ? 2 * j : 2 * j + 8) + hi * 8;
            v2h p0 = *(const v2h*)&Bh[b * DF + k0];
            v2h p1 = *(const v2h*)&Bh[b * DF + 32 + k0];
            b0[2 * j] = p0.x; b0[2 * j + 1] = p0.y;
            b1[2 * j] = p1.x; b1[2 * j + 1] = p1.y;
        }
        // C init: fold the rank-1 distance terms into the accumulator.
        const float sqb = sqB[b], db = dgB[b];
        v8f c;
        #pragma unroll
        for (int i = 0; i < 8; ++i) {
            const int arow = rowBase + i + hi * 8;
            const float dd = dgA[arow] - db;
            c[i] = -0.5f * (sqA[arow] + sqb + dd * dd);
        }
        v8f d;
        d = __builtin_amdgcn_wmma_f32_16x16x32_f16(false, a0, false, b0, (short)0, c, false, false);
        d = __builtin_amdgcn_wmma_f32_16x16x32_f16(false, a1, false, b1, (short)0, d, false, false);

        // Epilogue: pure maxes (exp deferred). Row maxes stay in registers.
        float cm = d[0];
        #pragma unroll
        for (int i = 0; i < 8; ++i) {
            rm[i] = fmaxf(rm[i], d[i]);
            if (i) cm = fmaxf(cm, d[i]);
        }
        cm = fmaxf(cm, __shfl_xor(cm, 16));             // combine both 8-row halves
        atomicMax(&colKey[b], fkey(cm));                // 16 distinct banks, no conflict
    }

    // Row maxima: reduce within each 16-lane half (rows rowBase+i and rowBase+i+8).
    float acc = 0.0f;
    #pragma unroll
    for (int i = 0; i < 8; ++i) {
        float v = rm[i];
        v = fmaxf(v, __shfl_xor(v, 1));
        v = fmaxf(v, __shfl_xor(v, 2));
        v = fmaxf(v, __shfl_xor(v, 4));
        v = fmaxf(v, __shfl_xor(v, 8));
        if (ln == 0) acc += __expf(2.0f * v);           // lanes 0 & 16 hold distinct rows
    }
    #pragma unroll
    for (int off = 16; off > 0; off >>= 1) acc += __shfl_xor(acc, off);
    if (lane == 0) atomicAdd(&mAcc[0], acc);

    __syncthreads();
    if (tid < NPG) atomicAdd(&mAcc[1], __expf(2.0f * fdec(colKey[tid])));
    __syncthreads();
    if (tid == 0) matchbuf[g * NGRAPH + h] = 0.5f * (mAcc[0] + mAcc[1]);
}

__global__ void finalize_kernel(const float* __restrict__ matchbuf,
                                const float* __restrict__ lam_raw,
                                float* __restrict__ out) {
    int i = blockIdx.x * blockDim.x + threadIdx.x;
    if (i < NGRAPH * NGRAPH) {
        const int g = i / NGRAPH, h = i % NGRAPH;
        const float lam = log1pf(__expf(lam_raw[0]));   // softplus
        const float m = 0.5f * (matchbuf[g * NGRAPH + h] + matchbuf[h * NGRAPH + g]);
        out[i] = (g == h) ? 0.0f : lam * ((float)NPG - m);
    }
}

extern "C" void kernel_launch(void* const* d_in, const int* in_sizes, int n_in,
                              void* d_out, int out_size, void* d_ws, size_t ws_size,
                              hipStream_t stream) {
    const float* x       = (const float*)d_in[0];   // [8192, 64] f32
    const int*   ei      = (const int*)d_in[1];     // [2, 131072]
    // d_in[2] (batch) and d_in[3] (edge_attr) are unused by the reference math.
    const float* lam_raw = (const float*)d_in[4];   // scalar
    float* out = (float*)d_out;                     // [64, 64] f32

    float* deg      = (float*)d_ws;                 // NNODE floats
    float* matchbuf = deg + NNODE;                  // G*G floats

    zero_deg_kernel<<<(NNODE + 255) / 256, 256, 0, stream>>>(deg);
    accum_deg_kernel<<<(NEDGE + 255) / 256, 256, 0, stream>>>(ei, deg);
    mcs_pair_kernel<<<dim3(NGRAPH, NGRAPH), 256, 0, stream>>>(x, deg, matchbuf);
    finalize_kernel<<<(NGRAPH * NGRAPH + 255) / 256, 256, 0, stream>>>(matchbuf, lam_raw, out);
}
